// MoEMixer_66949950210414
// MI455X (gfx1250) — matrix operand order repaired and checked by reading, not compiled, not run on hardware
//
#include <hip/hip_runtime.h>
#include <hip/hip_bf16.h>
#include <math.h>

// Problem sizes (fixed by reference)
#define BB 32
#define SS 1024
#define DD 512
#define EE 8
#define FF 2048
#define KSZ 5

// ffn tiling
#define MT 32          // rows per block tile
#define FC 64          // F-chunk width

typedef __bf16 bf16_t;
typedef __bf16 v16bf __attribute__((ext_vector_type(16)));
typedef float  v8f   __attribute__((ext_vector_type(8)));
typedef float  v4f   __attribute__((ext_vector_type(4)));
typedef unsigned int v4u __attribute__((ext_vector_type(4)));

union AF { v16bf v; v4u q[2]; };        // one WMMA 16x32 bf16 fragment (32B/lane)
union BfPack { bf16_t h[16]; v4u q[2]; };

__device__ __forceinline__ float gelu_exact(float v) {
  return 0.5f * v * (1.0f + erff(v * 0.70710678118654752f));
}
__device__ __forceinline__ float waveSum(float v) {
#pragma unroll
  for (int off = 16; off > 0; off >>= 1) v += __shfl_xor(v, off, 32);
  return v;
}

// ---------------------------------------------------------------------------
// Kernel 0: convert + transpose expert weights to bf16 for WMMA B-fragments.
// w1T[e][f][d] = bf16(w1[e][d][f])   (GEMM1 B: N=f rows, K=d contiguous)
// w2T[e][d][f] = bf16(w2[e][f][d])   (GEMM2 B: N=d rows, K=f contiguous)
// ---------------------------------------------------------------------------
__global__ __launch_bounds__(256) void prep_weights(
    const float* __restrict__ w1, const float* __restrict__ w2,
    bf16_t* __restrict__ w1T, bf16_t* __restrict__ w2T) {
  const size_t n = (size_t)EE * DD * FF;
  for (size_t i = (size_t)blockIdx.x * blockDim.x + threadIdx.x; i < n;
       i += (size_t)gridDim.x * blockDim.x) {
    size_t e = i / ((size_t)DD * FF);
    size_t rem = i % ((size_t)DD * FF);
    size_t f = rem / DD, d = rem % DD;          // w1T index decomposition
    w1T[i] = (bf16_t)w1[(e * DD + d) * FF + f];
    size_t d2 = rem / FF, f2 = rem % FF;        // w2T index decomposition
    w2T[i] = (bf16_t)w2[(e * FF + f2) * DD + d2];
  }
}

// ---------------------------------------------------------------------------
// Kernel 1: router. One block per batch. Pool -> LN -> GELU MLP -> top2+softmax
// ---------------------------------------------------------------------------
__global__ __launch_bounds__(256) void router_kernel(
    const float* __restrict__ x, const float* __restrict__ mask,
    const float* __restrict__ gln_g, const float* __restrict__ gln_b,
    const float* __restrict__ gw1, const float* __restrict__ gb1,
    const float* __restrict__ gw2, const float* __restrict__ gb2,
    int* __restrict__ topi, float* __restrict__ wexp) {
  __shared__ float sg[DD], sh1[DD], sh2[DD], part[256], slog[8];
  const int b = blockIdx.x, tid = threadIdx.x;

  // denom = clip(sum(mask), 1, inf)
  float p = 0.f;
  for (int s = tid; s < SS; s += 256) p += mask[b * SS + s];
  part[tid] = p; __syncthreads();
  for (int off = 128; off; off >>= 1) { if (tid < off) part[tid] += part[tid + off]; __syncthreads(); }
  const float denom = fmaxf(part[0], 1.0f);
  __syncthreads();

  // pooled g (each thread owns 2 columns)
  for (int c = tid; c < DD; c += 256) {
    float acc = 0.f;
    for (int s = 0; s < SS; ++s) acc += x[((size_t)b * SS + s) * DD + c] * mask[b * SS + s];
    sg[c] = acc / denom;
  }
  __syncthreads();

  // LayerNorm of pooled vector
  p = 0.f;
  for (int c = tid; c < DD; c += 256) p += sg[c];
  part[tid] = p; __syncthreads();
  for (int off = 128; off; off >>= 1) { if (tid < off) part[tid] += part[tid + off]; __syncthreads(); }
  const float mean = part[0] / DD;
  __syncthreads();
  p = 0.f;
  for (int c = tid; c < DD; c += 256) { float d = sg[c] - mean; p += d * d; }
  part[tid] = p; __syncthreads();
  for (int off = 128; off; off >>= 1) { if (tid < off) part[tid] += part[tid + off]; __syncthreads(); }
  const float rs = rsqrtf(part[0] / DD + 1e-5f);
  __syncthreads();
  for (int c = tid; c < DD; c += 256) sh1[c] = (sg[c] - mean) * rs * gln_g[c] + gln_b[c];
  __syncthreads();

  // MLP layer 1 (D x D) + exact GELU
  for (int f = tid; f < DD; f += 256) {
    float z = gb1[f];
    for (int d = 0; d < DD; ++d) z += sh1[d] * gw1[d * DD + f];
    sh2[f] = gelu_exact(z);
  }
  __syncthreads();

  // MLP layer 2 (D x 8)
  if (tid < 8) {
    float z = gb2[tid];
    for (int d = 0; d < DD; ++d) z += sh2[d] * gw2[d * 8 + tid];
    slog[tid] = z;
  }
  __syncthreads();

  if (tid == 0) {
    int i0 = 0;
    for (int j = 1; j < 8; ++j) if (slog[j] > slog[i0]) i0 = j;
    int i1 = (i0 == 0) ? 1 : 0;
    for (int j = 0; j < 8; ++j) if (j != i0 && slog[j] > slog[i1]) i1 = j;
    float w0 = 1.0f / (1.0f + expf(slog[i1] - slog[i0]));
    topi[b * 2 + 0] = i0; topi[b * 2 + 1] = i1;
    wexp[b * 2 + 0] = w0; wexp[b * 2 + 1] = 1.0f - w0;
  }
}

// ---------------------------------------------------------------------------
// Kernel 2a: hn = LN(x) with expert params; wave-per-row (8 rows/block)
// ---------------------------------------------------------------------------
__global__ __launch_bounds__(256) void ln1_kernel(
    const float* __restrict__ x, const int* __restrict__ topi,
    const float* __restrict__ ln_g, const float* __restrict__ ln_b,
    float* __restrict__ hn) {
  const int gw = blockIdx.x * 8 + (threadIdx.x >> 5);
  const int lane = threadIdx.x & 31;
  const int s = gw & (SS - 1);
  const int bk = gw >> 10;
  const int b = bk >> 1;
  const int e = topi[bk];
  const float* xr = x + ((size_t)b * SS + s) * DD;
  const int c0 = lane * 16;
  v4f xv[4];
#pragma unroll
  for (int i = 0; i < 4; ++i) xv[i] = *(const v4f*)(xr + c0 + 4 * i);
  float sm = 0.f, sq = 0.f;
#pragma unroll
  for (int i = 0; i < 4; ++i)
#pragma unroll
    for (int j = 0; j < 4; ++j) { float t = xv[i][j]; sm += t; sq += t * t; }
  sm = waveSum(sm) / DD;
  sq = waveSum(sq) / DD;
  const float rs = rsqrtf(fmaxf(sq - sm * sm, 0.0f) + 1e-5f);
  float* hr = hn + ((size_t)bk * SS + s) * DD + c0;
#pragma unroll
  for (int i = 0; i < 4; ++i) {
    v4f o;
#pragma unroll
    for (int j = 0; j < 4; ++j) {
      int d = c0 + 4 * i + j;
      o[j] = (xv[i][j] - sm) * rs * ln_g[e * DD + d] + ln_b[e * DD + d];
    }
    *(v4f*)(hr + 4 * i) = o;
  }
}

// ---------------------------------------------------------------------------
// Kernel 2b: depthwise conv(K=5) + residual -> y (f32); yn = LN(y) -> bf16
// ---------------------------------------------------------------------------
__global__ __launch_bounds__(256) void conv_ln2_kernel(
    const float* __restrict__ x, const int* __restrict__ topi,
    const float* __restrict__ conv_w, const float* __restrict__ conv_b,
    const float* __restrict__ ln_g, const float* __restrict__ ln_b,
    const float* __restrict__ hn, float* __restrict__ y_ws,
    bf16_t* __restrict__ yn_ws) {
  const int gw = blockIdx.x * 8 + (threadIdx.x >> 5);
  const int lane = threadIdx.x & 31;
  const int s = gw & (SS - 1);
  const int bk = gw >> 10;
  const int b = bk >> 1;
  const int e = topi[bk];
  const int c0 = lane * 16;
  const float* xr = x + ((size_t)b * SS + s) * DD;

  float yv[16];
#pragma unroll
  for (int i = 0; i < 4; ++i) {
    v4f xv = *(const v4f*)(xr + c0 + 4 * i);
#pragma unroll
    for (int j = 0; j < 4; ++j) yv[4 * i + j] = xv[j];
  }
#pragma unroll
  for (int t = 0; t < 16; ++t) {
    const int d = c0 + t;
    float acc = conv_b[e * DD + d];
#pragma unroll
    for (int j = 0; j < KSZ; ++j) {
      const int ss = s + j - 2;
      if (ss >= 0 && ss < SS)
        acc += hn[((size_t)bk * SS + ss) * DD + d] * conv_w[(e * DD + d) * KSZ + j];
    }
    yv[t] += acc;
  }
  float sm = 0.f, sq = 0.f;
#pragma unroll
  for (int t = 0; t < 16; ++t) { sm += yv[t]; sq += yv[t] * yv[t]; }
  sm = waveSum(sm) / DD;
  sq = waveSum(sq) / DD;
  const float rs = rsqrtf(fmaxf(sq - sm * sm, 0.0f) + 1e-5f);

  float* yr = y_ws + ((size_t)bk * SS + s) * DD + c0;
  BfPack pk;
#pragma unroll
  for (int i = 0; i < 4; ++i) {
    v4f o;
#pragma unroll
    for (int j = 0; j < 4; ++j) {
      const int t = 4 * i + j, d = c0 + t;
      o[j] = yv[t];
      pk.h[t] = (bf16_t)((yv[t] - sm) * rs * ln_g[e * DD + d] + ln_b[e * DD + d]);
    }
    *(v4f*)(yr + 4 * i) = o;
  }
  v4u* ynq = (v4u*)(yn_ws + ((size_t)bk * SS + s) * DD + c0);
  ynq[0] = pk.q[0];
  ynq[1] = pk.q[1];
}

// ---------------------------------------------------------------------------
// Kernel 3: fused expert FFN with BF16 WMMA.
// Block = (32 rows of S) x (all 512 cols), 8 waves, loops k in {0,1}.
// yn tile (32x512 bf16 = 32KB) staged in LDS once per k.
// Per 64-wide F-chunk:
//   GEMM1: wave -> one 16x16 ff1 tile over K=512 (16 wmma), A from LDS,
//          B (W1T) software-pipelined from global.
//   +b1, exact GELU, bf16 -> ping-pong 4KB LDS ff1 tile (ONE barrier/chunk:
//   the write of chunk i+2 into buffer p is ordered behind chunk i's reads
//   by chunk i+1's barrier).
//   GEMM2: acc[2][4] += ff1(32x64) @ W2T slab (16 wmma per wave).
// Epilogue: out = sum_k wgt_k*(y_k + ff_k + b2)*mask, then *mask.
// ---------------------------------------------------------------------------
__global__ __launch_bounds__(256) void ffn_kernel(
    const float* __restrict__ mask,
    const float* __restrict__ b1, const float* __restrict__ b2,
    const int* __restrict__ topi, const float* __restrict__ wexp,
    const bf16_t* __restrict__ w1T, const bf16_t* __restrict__ w2T,
    const float* __restrict__ y_ws, const bf16_t* __restrict__ yn_ws,
    float* __restrict__ out) {
  __shared__ bf16_t sYn[MT * DD];       // 32 KB staged activations
  __shared__ bf16_t sFf[2][MT * FC];    // 2 x 4 KB ping-pong ff1 chunk tile

  const int b = blockIdx.y;
  const int s0 = blockIdx.x * MT;
  const int tid = threadIdx.x;
  const int wave = tid >> 5;
  const int lane = tid & 31;
  const int hf = lane >> 4;         // lane half selects K offset
  const int l15 = lane & 15;

  const int m_t1 = wave >> 2;       // GEMM1: wave -> (m tile 0..1, f tile 0..3)
  const int f_t1 = wave & 3;

  for (int k = 0; k < 2; ++k) {
    const int e = topi[b * 2 + k];
    const float wgt = wexp[b * 2 + k];
    const bf16_t* ynP = yn_ws + (((size_t)(b * 2 + k)) * SS + s0) * DD;
    const float* yP = y_ws + (((size_t)(b * 2 + k)) * SS + s0) * DD;
    const bf16_t* w1e = w1T + (size_t)e * FF * DD;
    const bf16_t* w2e = w2T + (size_t)e * DD * FF;
    const float* b1e = b1 + e * FF;
    const float* b2e = b2 + e * DD;

    // ---- stage yn tile into LDS (contiguous 32KB copy) ----
    __syncthreads();  // previous k's LDS consumers done
    {
      const v4u* src = (const v4u*)ynP;
      v4u* dst = (v4u*)sYn;
#pragma unroll
      for (int i = 0; i < (MT * DD) / (8 * 256); ++i)
        dst[tid + i * 256] = src[tid + i * 256];
    }
    __syncthreads();

    v8f acc[2][4];
#pragma unroll
    for (int mt = 0; mt < 2; ++mt)
#pragma unroll
      for (int nt = 0; nt < 4; ++nt) acc[mt][nt] = v8f{};

    const bf16_t* aRow = sYn + (size_t)(m_t1 * 16 + l15) * DD;  // LDS A row

    int pp = 0;  // ping-pong parity
    for (int fc = 0; fc < FF; fc += FC, pp ^= 1) {
      // ---- GEMM1: c1 = yn(16x512) @ W1(512x16), B pipelined from global ----
      v8f c1 = v8f{};
      const bf16_t* bRow = w1e + (size_t)(fc + f_t1 * 16 + l15) * DD;
      AF Bcur;
      Bcur.q[0] = *(const v4u*)(bRow + hf * 8);
      Bcur.q[1] = *(const v4u*)(bRow + hf * 8 + 16);
#pragma unroll
      for (int kc = 0; kc < DD; kc += 32) {
        AF Bnxt;
        if (kc + 32 < DD) {
          Bnxt.q[0] = *(const v4u*)(bRow + kc + 32 + hf * 8);
          Bnxt.q[1] = *(const v4u*)(bRow + kc + 32 + hf * 8 + 16);
        }
        AF A;
        const bf16_t* ap = aRow + kc + hf * 8;
        A.q[0] = *(const v4u*)ap;
        A.q[1] = *(const v4u*)(ap + 16);
        c1 = __builtin_amdgcn_wmma_f32_16x16x32_bf16(false, A.v, false, Bcur.v,
                                                     (short)0, c1, false, false);
        if (kc + 32 < DD) Bcur = Bnxt;
      }
      const float b1v = b1e[fc + f_t1 * 16 + l15];
      bf16_t* sF = sFf[pp];
#pragma unroll
      for (int r = 0; r < 8; ++r) {
        const int row = m_t1 * 16 + r + hf * 8;
        sF[row * FC + f_t1 * 16 + l15] = (bf16_t)gelu_exact(c1[r] + b1v);
      }
      __syncthreads();  // single barrier per chunk (ping-pong ordering)

      // ---- GEMM2: acc += ff1(32x64) @ W2(64x512-slab), two K=32 halves ----
#pragma unroll
      for (int ko = 0; ko < 2; ++ko) {
        AF Bf[4];
#pragma unroll
        for (int nt = 0; nt < 4; ++nt) {
          const bf16_t* p =
              w2e + (size_t)(wave * 64 + nt * 16 + l15) * FF + fc + ko * 32 + hf * 8;
          Bf[nt].q[0] = *(const v4u*)p;
          Bf[nt].q[1] = *(const v4u*)(p + 16);
        }
#pragma unroll
        for (int mt = 0; mt < 2; ++mt) {
          AF A;
          const bf16_t* ap = sF + (mt * 16 + l15) * FC + ko * 32 + hf * 8;
          A.q[0] = *(const v4u*)ap;
          A.q[1] = *(const v4u*)(ap + 16);
#pragma unroll
          for (int nt = 0; nt < 4; ++nt)
            acc[mt][nt] = __builtin_amdgcn_wmma_f32_16x16x32_bf16(
                false, A.v, false, Bf[nt].v, (short)0, acc[mt][nt], false, false);
        }
      }
    }

    // ---- Epilogue: weighted residual combine (k=0 overwrite, k=1 RMW) ----
#pragma unroll
    for (int mt = 0; mt < 2; ++mt)
#pragma unroll
      for (int nt = 0; nt < 4; ++nt) {
        const int col = wave * 64 + nt * 16 + l15;
        const float b2v = b2e[col];
#pragma unroll
        for (int r = 0; r < 8; ++r) {
          const int row = mt * 16 + r + hf * 8;
          const float mrow = mask[b * SS + s0 + row];
          const float yv = yP[(size_t)row * DD + col];
          const float val = wgt * ((yv + acc[mt][nt][r] + b2v) * mrow);
          const size_t oidx = ((size_t)b * SS + s0 + row) * DD + col;
          if (k == 0) out[oidx] = val;
          else        out[oidx] = (out[oidx] + val) * mrow;  // final *mask
        }
      }
  }
}

// ---------------------------------------------------------------------------
// Host launcher. ws layout (bytes):
//   [0)      topi  : 32*2 int
//   [1024)   wexp  : 32*2 float
//   [4096)   w1T   : E*F*D bf16 (16.78 MB)
//   [+]      w2T   : E*D*F bf16 (16.78 MB)
//   [+]      hn    : B*2*S*D f32 (134.2 MB)
//   [+]      y     : B*2*S*D f32 (134.2 MB)
//   [+]      yn    : B*2*S*D bf16 (67.1 MB)      total ~369 MB
// ---------------------------------------------------------------------------
extern "C" void kernel_launch(void* const* d_in, const int* in_sizes, int n_in,
                              void* d_out, int out_size, void* d_ws, size_t ws_size,
                              hipStream_t stream) {
  (void)in_sizes; (void)n_in; (void)out_size; (void)ws_size;
  const float* x      = (const float*)d_in[0];
  const float* mask   = (const float*)d_in[1];
  const float* ln_g   = (const float*)d_in[2];
  const float* ln_b   = (const float*)d_in[3];
  const float* conv_w = (const float*)d_in[4];
  const float* conv_b = (const float*)d_in[5];
  const float* w1     = (const float*)d_in[6];
  const float* b1     = (const float*)d_in[7];
  const float* w2     = (const float*)d_in[8];
  const float* b2     = (const float*)d_in[9];
  const float* gln_g  = (const float*)d_in[10];
  const float* gln_b  = (const float*)d_in[11];
  const float* gw1    = (const float*)d_in[12];
  const float* gb1    = (const float*)d_in[13];
  const float* gw2    = (const float*)d_in[14];
  const float* gb2    = (const float*)d_in[15];
  float* out = (float*)d_out;

  char* ws = (char*)d_ws;
  const size_t szW = (size_t)EE * DD * FF * sizeof(bf16_t);   // 16.78 MB
  const size_t szA = (size_t)BB * 2 * SS * DD;                // act elements
  int*    topi = (int*)(ws + 0);
  float*  wexp = (float*)(ws + 1024);
  bf16_t* w1T  = (bf16_t*)(ws + 4096);
  bf16_t* w2T  = (bf16_t*)(ws + 4096 + szW);
  float*  hn   = (float*)(ws + 4096 + 2 * szW);
  float*  y_ws = (float*)(ws + 4096 + 2 * szW + szA * 4);
  bf16_t* yn   = (bf16_t*)(ws + 4096 + 2 * szW + 2 * szA * 4);

  prep_weights<<<4096, 256, 0, stream>>>(w1, w2, w1T, w2T);
  router_kernel<<<BB, 256, 0, stream>>>(x, mask, gln_g, gln_b, gw1, gb1, gw2, gb2,
                                        topi, wexp);
  ln1_kernel<<<(BB * 2 * SS) / 8, 256, 0, stream>>>(x, topi, ln_g, ln_b, hn);
  conv_ln2_kernel<<<(BB * 2 * SS) / 8, 256, 0, stream>>>(x, topi, conv_w, conv_b,
                                                         ln_g, ln_b, hn, y_ws, yn);
  ffn_kernel<<<dim3(SS / MT, BB), 256, 0, stream>>>(mask, b1, b2, topi, wexp,
                                                    w1T, w2T, y_ws, yn, out);
}
